// MultiLayer_LSTM_48369921688066
// MI455X (gfx1250) — matrix operand order repaired and checked by reading, not compiled, not run on hardware
//
#include <hip/hip_runtime.h>

typedef __attribute__((ext_vector_type(16))) _Float16 v16h;
typedef __attribute__((ext_vector_type(8)))  _Float16 v8h;
typedef __attribute__((ext_vector_type(8)))  float    v8f;

#define LAYERS 4
#define UDIM   512
#define BATCH  32
#define TSEQ   512
#define DDIM   512
#define GDIM   2048                   // 4*U (gate dim)
#define KT     16                     // K tiles of 32 (K = 512)
#define NFRAG  2048                   // (GDIM/16) * KT fragments per layer
#define FRAG_HALVES 512               // 32x16 f16 B-fragment
#define LAYER_W_ELEMS (DDIM * GDIM)   // 1,048,576

// ---------------- WMMA fragment loaders (per CDNA5 ISA layouts) ---

// A (16x32 f16): lanes 0-15 row M=lane, halves = K {0..7,16..23}; lanes 16-31
// row M=lane-16, halves = K {8..15,24..31}.  base points at (row 0, kBase).
__device__ inline v16h load_a_frag_f32(const float* base, int ld, int lane) {
  const float* p = base + (lane & 15) * ld + ((lane >> 4) * 8);
  v16h a;
#pragma unroll
  for (int i = 0; i < 8; ++i) { a[i] = (_Float16)p[i]; a[i + 8] = (_Float16)p[i + 16]; }
  return a;
}

__device__ inline v16h load_a_frag_f16(const _Float16* base, int ld, int lane) {
  const _Float16* p = base + (lane & 15) * ld + ((lane >> 4) * 8);
  v8h lo = *(const v8h*)p;
  v8h hi = *(const v8h*)(p + 16);
  v16h a;
#pragma unroll
  for (int i = 0; i < 8; ++i) { a[i] = lo[i]; a[i + 8] = hi[i]; }
  return a;
}

// ---------------- weight swizzle: fp32 [K x N] -> f16 B-fragments ------------
// B (32x16 f16): lanes 0-15: N=lane, halves = K {0..7,16..23}; lanes 16-31: N=lane-16,
// halves = K {8..15,24..31}.  Fragments stored [nt][kt][lane][16] so GEMM inner loop
// does one contiguous 32B load per fragment.
__global__ void swizzle_weights(const float* __restrict__ src, _Float16* __restrict__ dst) {
  int lane = threadIdx.x & 31;
  int gw = blockIdx.x * (blockDim.x >> 5) + (threadIdx.x >> 5);
  int layer = gw / NFRAG;
  int f = gw % NFRAG;
  int nt = f / KT;
  int kt = f % KT;
  const float* s = src + (size_t)layer * LAYER_W_ELEMS;
  _Float16* d = dst + (size_t)layer * LAYER_W_ELEMS + (size_t)f * FRAG_HALVES + lane * 16;
  int n = nt * 16 + (lane & 15);
  int kBase = kt * 32 + ((lane >> 4) * 8);
#pragma unroll
  for (int h = 0; h < 16; ++h)
    d[h] = (_Float16)s[(size_t)(kBase + h + ((h >> 3) * 8)) * GDIM + n];
}

// ---------------- input projection: xW = X @ W + b  (16384 x 512 x 2048) -----
// One wave per 16x64 output strip (1 M-tile x 4 N-tiles), K-chained WMMAs.
__global__ void proj_kernel(const float* __restrict__ X, const _Float16* __restrict__ Wf,
                            const float* __restrict__ bias, float* __restrict__ xw) {
  int lane = threadIdx.x & 31;
  int gw = blockIdx.x * 8 + (threadIdx.x >> 5);
  int mt = gw >> 5;   // 0..1023  (M tiles of 16)
  int ng = gw & 31;   // 0..31    (groups of 4 N-tiles)
  const float* Xrow = X + (size_t)(mt * 16) * DDIM;

  v8f acc[4];
#pragma unroll
  for (int j = 0; j < 4; ++j) {
    float bv = bias[(ng * 4 + j) * 16 + (lane & 15)];
#pragma unroll
    for (int r = 0; r < 8; ++r) acc[j][r] = bv;
  }
  for (int kt = 0; kt < KT; ++kt) {
    v16h a = load_a_frag_f32(Xrow + kt * 32, DDIM, lane);
#pragma unroll
    for (int j = 0; j < 4; ++j) {
      int ntg = ng * 4 + j;
      const v16h* bp = (const v16h*)(Wf + (size_t)(ntg * KT + kt) * FRAG_HALVES + lane * 16);
      acc[j] = __builtin_amdgcn_wmma_f32_16x16x32_f16(false, a, false, *bp,
                                                      (short)0, acc[j], false, false);
    }
  }
  // C/D layout: VGPR r -> M = r (+8 for lanes 16-31), N = lane&15
  int rowOff = mt * 16 + ((lane < 16) ? 0 : 8);
#pragma unroll
  for (int j = 0; j < 4; ++j) {
    int col = (ng * 4 + j) * 16 + (lane & 15);
#pragma unroll
    for (int r = 0; r < 8; ++r)
      xw[(size_t)(rowOff + r) * GDIM + col] = acc[j][r];
  }
}

// ---------------- inter-workgroup barrier (16 persistent WGs) ----------------
__device__ inline void grid_sync(unsigned* bar, unsigned target) {
  __builtin_amdgcn_fence(__ATOMIC_RELEASE, "agent");   // publish my h stores
  __syncthreads();
  if (threadIdx.x == 0) {
    __hip_atomic_fetch_add(bar, 1u, __ATOMIC_RELAXED, __HIP_MEMORY_SCOPE_AGENT);
    while (__hip_atomic_load(bar, __ATOMIC_RELAXED, __HIP_MEMORY_SCOPE_AGENT) < target)
      __builtin_amdgcn_s_sleep(8);
  }
  __syncthreads();
  __builtin_amdgcn_fence(__ATOMIC_ACQUIRE, "agent");   // see everyone's h
}

__global__ void init_bar(unsigned* bar) { if (threadIdx.x == 0) *bar = 0u; }

// ---------------- recurrent scan (persistent, 16 blocks x 256 threads) -------
// Block w owns U-columns [w*32, w*32+32).  Per step: async-stage the shared
// hidden state h (32KB f16) from global into LDS (CDNA5 ASYNCcnt path), run
// 16 chained WMMAs per z-tile against LDS-resident Uh fragments, apply gates,
// publish new h through L2 for the next step.
__global__ void scan_kernel(const float* __restrict__ xw, const _Float16* __restrict__ Uf,
                            float* __restrict__ hs, float* __restrict__ sh,
                            float* __restrict__ sc, _Float16* __restrict__ h16,
                            unsigned* __restrict__ bar, float zz) {
  extern __shared__ char smem[];
  _Float16* uf   = (_Float16*)smem;                    // 8 ntg x 16 kt x 512 halves = 128 KB
  _Float16* hlds = (_Float16*)(smem + 131072);         // h stage [32][512] f16   =  32 KB
  float* zbuf = (float*)(smem + 131072 + 32768);       // [4][32][32] gate exchange = 16 KB
  float* cbuf = zbuf + 4096;                           // [32][32] cell state
  float* hbuf = cbuf + 1024;                           // [32][32] hidden state

  const int tid  = threadIdx.x;
  const int lane = tid & 31;
  const int q    = tid >> 5;      // wave 0..7
  const int w    = blockIdx.x;    // 0..15

  // Preload this block's slice of Uh into LDS (CDNA5: 320KB/WGP).
  for (int j = 0; j < 8; ++j) {
    int gate = j >> 1, ntile = j & 1;
    int ntg = gate * 32 + w * 2 + ntile;
    const uint4* s = (const uint4*)(Uf + (size_t)ntg * KT * FRAG_HALVES);
    uint4* d = (uint4*)(uf + j * KT * FRAG_HALVES);
    for (int i = tid; i < KT * FRAG_HALVES / 8; i += 256) d[i] = s[i];
  }
  for (int i = tid; i < 1024; i += 256) { cbuf[i] = 0.f; hbuf[i] = 0.f; }
  for (int i = tid; i < 1024; i += 256)
    h16[(i >> 5) * UDIM + w * 32 + (i & 31)] = (_Float16)0.f;

  // LDS byte offset of the h staging buffer (low 32 bits of the pointer).
  const unsigned hldsBase = (unsigned)(uintptr_t)hlds;

  unsigned barNum = 0;
  grid_sync(bar, ++barNum * 16);   // all h16 slices zeroed & LDS loaded

  for (int t = 0; t < TSEQ; ++t) {
    // --- async-stage full h (32KB) global -> LDS; tracked by ASYNCcnt -------
#pragma unroll
    for (int i = 0; i < 8; ++i) {
      int idx = tid + i * 256;                       // 2048 x b128 chunks
      unsigned ldso = hldsBase + (unsigned)(idx * 16);
      const void* g = (const char*)h16 + (size_t)idx * 16;
      asm volatile("global_load_async_to_lds_b128 %0, %1, off"
                   :: "v"(ldso), "v"(g) : "memory");
    }
    asm volatile("s_wait_asynccnt 0x0" ::: "memory");
    __syncthreads();                                 // all waves' chunks in LDS

    // --- z = xW[:,t,cols] + h @ Uh[:,cols] : 16 tiles, 2 per wave -----------
#pragma unroll
    for (int s2 = 0; s2 < 2; ++s2) {
      int tix = q * 2 + s2;
      int gate = tix >> 2, mtile = (tix >> 1) & 1, ntile = tix & 1;
      int j = gate * 2 + ntile;
      int zcol = gate * UDIM + w * 32 + ntile * 16 + (lane & 15);
      int rowOff = mtile * 16 + ((lane < 16) ? 0 : 8);
      v8f acc;
#pragma unroll
      for (int r = 0; r < 8; ++r)      // init accumulator from precomputed xW
        acc[r] = xw[((size_t)(rowOff + r) * TSEQ + t) * GDIM + zcol];
      if (t + 1 < TSEQ)                // hide next step's xW HBM latency
        __builtin_prefetch(&xw[((size_t)rowOff * TSEQ + (t + 1)) * GDIM + zcol], 0, 1);
      for (int kt = 0; kt < KT; ++kt) {
        v16h a = load_a_frag_f16(hlds + (mtile * 16) * UDIM + kt * 32, UDIM, lane);
        const v16h* bp = (const v16h*)(uf + (j * KT + kt) * FRAG_HALVES + lane * 16);
        acc = __builtin_amdgcn_wmma_f32_16x16x32_f16(false, a, false, *bp,
                                                     (short)0, acc, false, false);
      }
      int ncol = ntile * 16 + (lane & 15);
#pragma unroll
      for (int r = 0; r < 8; ++r)
        zbuf[(gate * 32 + rowOff + r) * 32 + ncol] = acc[r];
    }
    __syncthreads();
    // --- gates + state update (1024 elements, 4 per thread) -----------------
#pragma unroll
    for (int k = 0; k < 4; ++k) {
      int el = tid + k * 256;
      int b_ = el >> 5, col = el & 31;
      float zi = zbuf[(0 * 32 + b_) * 32 + col];
      float zf = zbuf[(1 * 32 + b_) * 32 + col];
      float zg = zbuf[(2 * 32 + b_) * 32 + col];
      float zo = zbuf[(3 * 32 + b_) * 32 + col];
      float ig = 1.f / (1.f + __expf(-zi));
      float fg = 1.f / (1.f + __expf(-zf));
      float gg = tanhf(zg);
      float og = 1.f / (1.f + __expf(-zo));
      float c = fg * cbuf[el] + ig * gg;
      float h = og * tanhf(c);
      cbuf[el] = c; hbuf[el] = h;
      int u = w * 32 + col;
      hs[((size_t)b_ * TSEQ + t) * UDIM + u] = h;   // raw h (LN comes later)
      h16[b_ * UDIM + u] = (_Float16)h;             // next step's A matrix
    }
    grid_sync(bar, ++barNum * 16);
  }
  // final states (+ zoneout scale for layers > 0; h0=c0=0)
#pragma unroll
  for (int k = 0; k < 4; ++k) {
    int el = tid + k * 256;
    int b_ = el >> 5, u = w * 32 + (el & 31);
    sh[b_ * UDIM + u] = zz * hbuf[el];
    sc[b_ * UDIM + u] = zz * cbuf[el];
  }
}

// ---------------- LayerNorm (in place, one wave per (b,t) row) ---------------
__global__ void ln_kernel(float* __restrict__ y, const float* __restrict__ gamma,
                          const float* __restrict__ beta) {
  int lane = threadIdx.x & 31;
  int row = blockIdx.x * 8 + (threadIdx.x >> 5);
  float* p = y + (size_t)row * UDIM;
  float x[16], s = 0.f;
#pragma unroll
  for (int i = 0; i < 16; ++i) { x[i] = p[lane + i * 32]; s += x[i]; }
#pragma unroll
  for (int m = 16; m >= 1; m >>= 1) s += __shfl_xor(s, m, 32);
  float mu = s * (1.f / UDIM);
  float v = 0.f;
#pragma unroll
  for (int i = 0; i < 16; ++i) { x[i] -= mu; v += x[i] * x[i]; }
#pragma unroll
  for (int m = 16; m >= 1; m >>= 1) v += __shfl_xor(v, m, 32);
  float rs = rsqrtf(v * (1.f / UDIM) + 1e-3f);
#pragma unroll
  for (int i = 0; i < 16; ++i) {
    int c = lane + i * 32;
    p[c] = x[i] * rs * gamma[c] + beta[c];
  }
}

// ---------------- host launcher ----------------------------------------------
extern "C" void kernel_launch(void* const* d_in, const int* in_sizes, int n_in,
                              void* d_out, int out_size, void* d_ws, size_t ws_size,
                              hipStream_t stream) {
  (void)in_sizes; (void)n_in; (void)out_size; (void)ws_size;
  const float* inputs = (const float*)d_in[0];   // [32,512,512]
  const float* W      = (const float*)d_in[1];   // [4,512,2048]
  const float* Uh     = (const float*)d_in[2];   // [4,512,2048]
  const float* bias   = (const float*)d_in[3];   // [4,2048]
  const float* gamma  = (const float*)d_in[4];   // [4,512]
  const float* beta   = (const float*)d_in[5];   // [4,512]

  float* all_out = (float*)d_out;                                   // [4,32,512,512]
  float* st_h = all_out + (size_t)LAYERS * BATCH * TSEQ * UDIM;     // [4,32,512]
  float* st_c = st_h + (size_t)LAYERS * BATCH * UDIM;               // [4,32,512]

  char* ws = (char*)d_ws;
  float* xw = (float*)ws;                                           // 128 MB
  size_t off = (size_t)BATCH * TSEQ * GDIM * sizeof(float);
  _Float16* Wf = (_Float16*)(ws + off); off += (size_t)LAYERS * LAYER_W_ELEMS * 2;
  _Float16* Uf = (_Float16*)(ws + off); off += (size_t)LAYERS * LAYER_W_ELEMS * 2;
  _Float16* h16 = (_Float16*)(ws + off); off += (size_t)BATCH * UDIM * 2;
  unsigned* bar = (unsigned*)(ws + off);

  // one-time (per call) weight swizzle into WMMA B-fragment order
  swizzle_weights<<<LAYERS * NFRAG / 8, 256, 0, stream>>>(W, Wf);
  swizzle_weights<<<LAYERS * NFRAG / 8, 256, 0, stream>>>(Uh, Uf);

  const size_t smemScan = 131072 + 32768 + (4096 + 1024 + 1024) * sizeof(float); // 184 KB

  for (int l = 0; l < LAYERS; ++l) {
    const float* X = (l == 0) ? inputs
                              : (all_out + (size_t)(l - 1) * BATCH * TSEQ * UDIM);
    proj_kernel<<<4096, 256, 0, stream>>>(X, Wf + (size_t)l * LAYER_W_ELEMS,
                                          bias + l * GDIM, xw);
    init_bar<<<1, 64, 0, stream>>>(bar);
    scan_kernel<<<16, 256, smemScan, stream>>>(
        xw, Uf + (size_t)l * LAYER_W_ELEMS,
        all_out + (size_t)l * BATCH * TSEQ * UDIM,
        st_h + (size_t)l * BATCH * UDIM, st_c + (size_t)l * BATCH * UDIM,
        h16, bar, (l > 0) ? 0.9f : 1.0f);
    ln_kernel<<<BATCH * TSEQ / 8, 256, 0, stream>>>(
        all_out + (size_t)l * BATCH * TSEQ * UDIM, gamma + l * UDIM, beta + l * UDIM);
  }
}
